// WhileOpLSTM_86242943304254
// MI455X (gfx1250) — matrix operand order repaired and checked by reading, not compiled, not run on hardware
//
#include <hip/hip_runtime.h>
#include <hip/hip_bf16.h>
#include <stdint.h>
#include <math.h>

// ---------------- problem constants ----------------
#define S_LEN   2048
#define BATCH   32
#define DDIM    512
#define HDIM    512
#define NLAYERS 2
#define NCHUNK  16           // N-column chunks == WGs per batch-half
#define CHUNK_COLS 32
#define HALVES  2

#define GATE_KT 32           // 1024 / 32
#define OUT_KT  16           // 512  / 32
#define TILES_PER_CHUNK (NLAYERS*GATE_KT*2 + NLAYERS*OUT_KT*2)   // 128 + 64 = 192
#define TILE_USHORTS 512     // 32x16 bf16 tile
#define CHUNK_USHORTS (TILES_PER_CHUNK*TILE_USHORTS)             // 98304
#define WEIGHTS_USHORTS ((size_t)NCHUNK*CHUNK_USHORTS)           // 1,572,864

// ---------------- d_ws layout (bytes) ----------------
#define WS_WEIGHTS 0
#define WS_HD   (WEIGHTS_USHORTS*2)                   // [2 half][2 parity][16*512] bf16
#define WS_X    (WS_HD + (size_t)HALVES*2*16*HDIM*2)  // [2 half][16*512] bf16
#define WS_BAR  (WS_X  + (size_t)HALVES*16*DDIM*2)    // uint cnt[2], gen[2]

#define XS_STRIDE 520        // padded LDS row stride (ushorts) to spread banks

typedef __attribute__((ext_vector_type(16))) __bf16 v16bf;
typedef __attribute__((ext_vector_type(8)))  float  v8f;

union ABReg {
    unsigned int u[8];
    uint4        q[2];
    v16bf        v;
};

__device__ inline unsigned short f2bf(float f) {
    unsigned int u = __float_as_uint(f);
    unsigned int r = u + 0x7FFFu + ((u >> 16) & 1u);   // round-to-nearest-even
    return (unsigned short)(r >> 16);
}

// ---- fast transcendentals on the recurrence critical path ------------------
// v_exp_f32 / v_rcp_f32 are branch-free hardware ops; v_tanh_f32 is new on
// gfx1250. Avoid the libm range-reduction + exec-mask slow paths entirely.
// Fallbacks must be valid in BOTH compilation passes (host + device), so the
// non-builtin paths use plain libm names.
#if __has_builtin(__builtin_amdgcn_exp2f)
#define EXP2F(x) __builtin_amdgcn_exp2f(x)
#else
#define EXP2F(x) exp2f(x)
#endif
#if __has_builtin(__builtin_amdgcn_rcpf)
#define RCPF(x) __builtin_amdgcn_rcpf(x)
#else
#define RCPF(x) (1.0f / (x))
#endif

#define LOG2E  1.442695041f
#define LOG2E2 2.885390082f

__device__ inline float fast_sigmoid(float x) {
    // 1 / (1 + 2^(-x*log2(e)))
    return RCPF(1.0f + EXP2F(-LOG2E * x));
}
__device__ inline float fast_tanh(float x) {
#if __has_builtin(__builtin_amdgcn_tanhf)
    return __builtin_amdgcn_tanhf(x);        // v_tanh_f32 (gfx1250)
#else
    // tanh(x) = 2/(1 + e^(-2x)) - 1
    return fmaf(2.0f, RCPF(1.0f + EXP2F(-LOG2E2 * x)), -1.0f);
#endif
}

// Sense-reversal barrier across the 16 WGs of one batch-half.
__device__ inline void half_barrier(unsigned int* cnt, unsigned int* gen) {
    __threadfence();
    __syncthreads();
    if (threadIdx.x == 0) {
        unsigned int g = __hip_atomic_load(gen, __ATOMIC_RELAXED, __HIP_MEMORY_SCOPE_AGENT);
        unsigned int a = __hip_atomic_fetch_add(cnt, 1u, __ATOMIC_ACQ_REL, __HIP_MEMORY_SCOPE_AGENT);
        if (a == NCHUNK - 1) {
            __hip_atomic_store(cnt, 0u, __ATOMIC_RELAXED, __HIP_MEMORY_SCOPE_AGENT);
            __hip_atomic_fetch_add(gen, 1u, __ATOMIC_RELEASE, __HIP_MEMORY_SCOPE_AGENT);
        } else {
            while (__hip_atomic_load(gen, __ATOMIC_ACQUIRE, __HIP_MEMORY_SCOPE_AGENT) == g)
                __builtin_amdgcn_s_sleep(2);
        }
    }
    __syncthreads();
    __threadfence();
}

// ---------------------------------------------------------------------------
// One-time prep: fp32 weights -> bf16 WMMA-B tiles, grouped per N-chunk so the
// persistent kernel copies one contiguous 192KB block into LDS.
// Tile layout (32x16): lane l holds N-column l%16, 16 K values in WMMA order
// (k = 8h+s for s<8, 16+8h+(s-8) for s>=8, h = l/16).
// ---------------------------------------------------------------------------
__global__ void prep_weights_kernel(const float* __restrict__ w_gate,
                                    const float* __restrict__ w_output,
                                    unsigned short* __restrict__ wTiles) {
    size_t idx = (size_t)blockIdx.x * blockDim.x + threadIdx.x;
    if (idx >= WEIGHTS_USHORTS) return;
    int c    = (int)(idx / CHUNK_USHORTS);
    int r    = (int)(idx % CHUNK_USHORTS);
    int tile = r / TILE_USHORTS;
    int u    = r % TILE_USHORTS;
    int K = ((u & 15) < 8) ? (((u >> 8) << 3) + (u & 15)) : (16 + ((u >> 8) << 3) + (u & 15) - 8);           // 0..31
    int N = (u >> 4) & 15;           // 0..15
    float v;
    if (tile < NLAYERS * GATE_KT * 2) {                 // gate tiles
        int d  = tile / (GATE_KT * 2);
        int r2 = tile % (GATE_KT * 2);
        int kt = r2 >> 1, nt = r2 & 1;
        int col = c * CHUNK_COLS + nt * 16 + N;
        v = w_gate[((size_t)d * (DDIM + HDIM) + kt * 32 + K) * HDIM + col];
    } else {                                            // output-projection tiles
        int t2 = tile - NLAYERS * GATE_KT * 2;
        int d  = t2 / (OUT_KT * 2);
        int r2 = t2 % (OUT_KT * 2);
        int kt = r2 >> 1, nt = r2 & 1;
        int col = c * CHUNK_COLS + nt * 16 + N;
        v = w_output[((size_t)d * HDIM + kt * 32 + K) * HDIM + col];
    }
    wTiles[idx] = f2bf(v);
}

__global__ void init_barrier_kernel(unsigned int* bar) {
    if (threadIdx.x < 4) bar[threadIdx.x] = 0u;
}

// ---------------------------------------------------------------------------
// Persistent LSTM kernel. Grid: 32 WGs = 2 batch-halves x 16 N-chunks.
// Block: 64 threads = 2 waves; each wave owns one 16x16 output tile and keeps
// its slice of the cell state in accumulator VGPRs for all 2048 steps.
// ---------------------------------------------------------------------------
__global__ void __launch_bounds__(64)
lstm_persistent_kernel(const float* __restrict__ input_seq,
                       const float* __restrict__ b_gate,
                       const float* __restrict__ b_output,
                       const unsigned short* __restrict__ wTilesG,
                       unsigned short* __restrict__ hdBuf,   // [2][2][16*HDIM] bf16
                       unsigned short* __restrict__ xBuf,    // [2][16*DDIM] bf16
                       unsigned int*  __restrict__ bar,      // cnt[2], gen[2]
                       float* __restrict__ out) {            // [S, B, H] fp32
    extern __shared__ unsigned char smemRaw[];
    unsigned short* wT = (unsigned short*)smemRaw;           // 98304 ushorts (192KB)
    unsigned short* Xs = wT + CHUNK_USHORTS;                 // 16 x XS_STRIDE
    unsigned short* Hs = Xs + 16 * XS_STRIDE;                // 16 x XS_STRIDE
    float* biasLds = (float*)(Hs + 16 * XS_STRIDE);          // [4][32]

    const int tid  = threadIdx.x;
    const int lane = tid & 31;
    const int wave = tid >> 5;            // 0..1 == nt (N-tile within chunk)
    const int h    = blockIdx.x >> 4;     // batch half
    const int c    = blockIdx.x & 15;     // N-chunk
    const int row0 = h * 16;
    const int n0   = c * CHUNK_COLS + wave * 16;  // global column base of my tile

    // ---- load this chunk's weight tiles into LDS (192KB, uint4 copies) ----
    {
        const uint4* src = (const uint4*)(wTilesG + (size_t)c * CHUNK_USHORTS);
        uint4* dst = (uint4*)wT;
        for (int i = tid; i < CHUNK_USHORTS / 8; i += 64) dst[i] = src[i];
    }
    if (tid < 32) {
        biasLds[0 * 32 + tid] = b_gate[0 * HDIM + c * CHUNK_COLS + tid];
        biasLds[1 * 32 + tid] = b_gate[1 * HDIM + c * CHUNK_COLS + tid];
        biasLds[2 * 32 + tid] = b_output[0 * HDIM + c * CHUNK_COLS + tid];
        biasLds[3 * 32 + tid] = b_output[1 * HDIM + c * CHUNK_COLS + tid];
    }
    __syncthreads();

    // Persistent cell state: C-layout element i -> (m = i + 8*(lane>=16), n = lane&15)
    v8f cell = {0.f, 0.f, 0.f, 0.f, 0.f, 0.f, 0.f, 0.f};

    const int m        = lane & 15;            // A-matrix row for this lane
    const int koffBase = (lane >> 4) * 8;      // A-layout K offset for upper lanes
    const int nloc     = lane & 15;            // C/D column within tile
    const int mbase    = (lane >> 4) * 8;      // C/D row base

    unsigned int p = 0;                        // hidden ping-pong parity
    unsigned int* cntP = &bar[h];
    unsigned int* genP = &bar[2 + h];
    unsigned short* XbufH = xBuf + (size_t)h * 16 * DDIM;

    for (int t = 0; t < S_LEN; ++t) {
        for (int d = 0; d < NLAYERS; ++d) {
            // ================= Phase A: stage x_t and hidden =================
            if (d == 0) {
                // input_seq [B,S,D] fp32 -> bf16 into Xs
                for (int i = tid; i < 16 * DDIM / 2; i += 64) {
                    int e = i * 2, mm = e >> 9, k = e & 511;
                    const float* sp = input_seq + ((size_t)(row0 + mm) * S_LEN + t) * DDIM + k;
                    unsigned int pk = (unsigned int)f2bf(sp[0]) |
                                      ((unsigned int)f2bf(sp[1]) << 16);
                    *(unsigned int*)(Xs + mm * XS_STRIDE + k) = pk;
                }
                // prefetch next timestep's input rows (global_prefetch_b8)
                if (t + 1 < S_LEN) {
                    for (int i = tid; i < 256; i += 64) {
                        int mm = i >> 4, seg = i & 15;   // 16 x 128B lines per row
                        __builtin_prefetch(
                            input_seq + ((size_t)(row0 + mm) * S_LEN + (t + 1)) * DDIM
                                      + seg * 32, 0, 1);
                    }
                }
            } else {
                const unsigned int* sp = (const unsigned int*)XbufH;
                for (int i = tid; i < 16 * DDIM / 2; i += 64) {
                    int e = i * 2, mm = e >> 9, k = e & 511;
                    *(unsigned int*)(Xs + mm * XS_STRIDE + k) = sp[i];
                }
            }
            if (t == 0 && d == 0) {
                for (int i = tid; i < 16 * HDIM / 2; i += 64) {
                    int e = i * 2, mm = e >> 9, k = e & 511;
                    *(unsigned int*)(Hs + mm * XS_STRIDE + k) = 0u;
                }
            } else {
                const unsigned int* sp =
                    (const unsigned int*)(hdBuf + ((size_t)h * 2 + p) * 16 * HDIM);
                for (int i = tid; i < 16 * HDIM / 2; i += 64) {
                    int e = i * 2, mm = e >> 9, k = e & 511;
                    *(unsigned int*)(Hs + mm * XS_STRIDE + k) = sp[i];
                }
            }
            __syncthreads();

            // ================= gate GEMM: [16,1024] x [1024,16] ==============
            float bg = biasLds[d * 32 + wave * 16 + nloc];
            v8f acc = {bg, bg, bg, bg, bg, bg, bg, bg};
            for (int kt = 0; kt < GATE_KT; ++kt) {
                const unsigned short* Abase =
                    (kt < 16) ? (Xs + m * XS_STRIDE + kt * 32)
                              : (Hs + m * XS_STRIDE + (kt - 16) * 32);
                ABReg au;
#pragma unroll
                for (int j = 0; j < 8; ++j) {
                    int off = koffBase + ((j < 4) ? 2 * j : 16 + 2 * (j - 4));
                    au.u[j] = *(const unsigned int*)(Abase + off);
                }
                ABReg bu;
                const uint4* bt = (const uint4*)(wT +
                    (size_t)((d * GATE_KT + kt) * 2 + wave) * TILE_USHORTS + lane * 16);
                bu.q[0] = bt[0];
                bu.q[1] = bt[1];
                acc = __builtin_amdgcn_wmma_f32_16x16x32_bf16(
                    false, au.v, false, bu.v, (short)0, acc, false, false);
            }

            // ================= activation + state update =====================
            v8f hidv;
#pragma unroll
            for (int i = 0; i < 8; ++i) {
                float g  = acc[i];
                float sg = fast_sigmoid(g);              // f = i = o (shared gate)
                float tg = fast_tanh(g);
                float cn = cell[i] * sg + tg * sg;
                cell[i]  = cn;
                hidv[i]  = fast_tanh(cn) * sg;
            }
            // publish new hidden chunk (bf16) to the ping-pong buffer
            {
                unsigned short* dst = hdBuf + ((size_t)h * 2 + (p ^ 1)) * 16 * HDIM;
#pragma unroll
                for (int r = 0; r < 8; ++r)
                    dst[(mbase + r) * HDIM + n0 + nloc] = f2bf(hidv[r]);
            }
            half_barrier(cntP, genP);

            // ================= Phase B: output projection ====================
            {
                const unsigned int* sp =
                    (const unsigned int*)(hdBuf + ((size_t)h * 2 + (p ^ 1)) * 16 * HDIM);
                for (int i = tid; i < 16 * HDIM / 2; i += 64) {
                    int e = i * 2, mm = e >> 9, k = e & 511;
                    *(unsigned int*)(Hs + mm * XS_STRIDE + k) = sp[i];
                }
            }
            __syncthreads();

            float bo = biasLds[(2 + d) * 32 + wave * 16 + nloc];
            v8f acc2 = {bo, bo, bo, bo, bo, bo, bo, bo};
            for (int kt = 0; kt < OUT_KT; ++kt) {
                const unsigned short* Abase = Hs + m * XS_STRIDE + kt * 32;
                ABReg au;
#pragma unroll
                for (int j = 0; j < 8; ++j) {
                    int off = koffBase + ((j < 4) ? 2 * j : 16 + 2 * (j - 4));
                    au.u[j] = *(const unsigned int*)(Abase + off);
                }
                ABReg bu;
                const uint4* bt = (const uint4*)(wT +
                    (size_t)(NLAYERS * GATE_KT * 2 +
                             (d * OUT_KT + kt) * 2 + wave) * TILE_USHORTS + lane * 16);
                bu.q[0] = bt[0];
                bu.q[1] = bt[1];
                acc2 = __builtin_amdgcn_wmma_f32_16x16x32_bf16(
                    false, au.v, false, bu.v, (short)0, acc2, false, false);
            }

            if (d == 0) {
                // x for next layer (bf16 all-gather buffer)
#pragma unroll
                for (int r = 0; r < 8; ++r)
                    XbufH[(mbase + r) * DDIM + n0 + nloc] = f2bf(acc2[r]);
            } else {
                // final layer output -> out[t, b, :] fp32
                float* dst = out + (size_t)t * BATCH * HDIM + (size_t)row0 * HDIM;
#pragma unroll
                for (int r = 0; r < 8; ++r)
                    dst[(mbase + r) * HDIM + n0 + nloc] = acc2[r];
            }
            half_barrier(cntP, genP);
            p ^= 1;
        }
    }
}

// ---------------------------------------------------------------------------
extern "C" void kernel_launch(void* const* d_in, const int* in_sizes, int n_in,
                              void* d_out, int out_size, void* d_ws, size_t ws_size,
                              hipStream_t stream) {
    (void)in_sizes; (void)n_in; (void)out_size; (void)ws_size;

    const float* input_seq = (const float*)d_in[0];
    const float* w_gate    = (const float*)d_in[1];
    const float* b_gate    = (const float*)d_in[2];
    const float* w_output  = (const float*)d_in[3];
    const float* b_output  = (const float*)d_in[4];

    unsigned short* wTiles = (unsigned short*)((char*)d_ws + WS_WEIGHTS);
    unsigned short* hdBuf  = (unsigned short*)((char*)d_ws + WS_HD);
    unsigned short* xBuf   = (unsigned short*)((char*)d_ws + WS_X);
    unsigned int*   bar    = (unsigned int*)((char*)d_ws + WS_BAR);

    // One-time (per call) weight retile + barrier reset; stream-ordered.
    prep_weights_kernel<<<(unsigned)((WEIGHTS_USHORTS + 255) / 256), 256, 0, stream>>>(
        w_gate, w_output, wTiles);
    init_barrier_kernel<<<1, 64, 0, stream>>>(bar);

    const size_t ldsBytes = (size_t)CHUNK_USHORTS * 2      // weights 192KB
                          + 2 * (16 * XS_STRIDE * 2)       // X/H staging ~32.5KB
                          + 4 * 32 * sizeof(float);        // biases
    (void)hipFuncSetAttribute((const void*)lstm_persistent_kernel,
                              hipFuncAttributeMaxDynamicSharedMemorySize,
                              (int)ldsBytes);

    lstm_persistent_kernel<<<HALVES * NCHUNK, 64, ldsBytes, stream>>>(
        input_seq, b_gate, b_output, wTiles, hdBuf, xBuf, bar, (float*)d_out);
}
